// CharRNN_29463475651457
// MI455X (gfx1250) — compile-verified, matrix operand order checked by reading, not run on hardware
//
#include <hip/hip_runtime.h>
#include <hip/hip_bf16.h>

#define B_ 256
#define S_ 256
#define H_ 1024
#define V_ 128

typedef __bf16 bf16_t;
typedef __attribute__((ext_vector_type(16))) __bf16        v16bf;
typedef __attribute__((ext_vector_type(8)))  float         v8f;
typedef __attribute__((ext_vector_type(4)))  unsigned int  v4u;

union FragU { v16bf v; v4u q[2]; };

// CDNA5 16-bit WMMA operand layout (ISA 7.12.2):
//   lane L holds row (L & 15); K-half base kb = (L<16) ? 0 : 8
//   elements 0..7  -> K = k0+kb+0 .. k0+kb+7     (one 16B load, offset 0)
//   elements 8..15 -> K = k0+kb+16 .. k0+kb+23   (one 16B load, offset +32B)
// NOTE: address expressions stay rooted at the kernel-arg pointer so clang's
// address-space inference emits global_load_b128 (not flat_load_b128, which
// would also tick DScnt/XCNT).
__device__ __forceinline__ v16bf load_frag(const bf16_t* __restrict__ base,
                                           int row, int k0, int ld, int half) {
  const bf16_t* p = base + (size_t)row * ld + k0 + half * 8;
  FragU f;
  f.q[0] = *(const v4u*)(p);
  f.q[1] = *(const v4u*)(p + 16);
  return f.v;
}

// acc[MT][NT] += A[m0:m0+MT*16, :K] * W[n0:n0+NT*16, :K]^T  (both row-major, K contiguous)
template <int MT, int NT>
__device__ __forceinline__ void gemm_bf16(const bf16_t* __restrict__ A, int ldA,
                                          const bf16_t* __restrict__ W, int ldW,
                                          int K, int m0, int n0, v8f acc[MT][NT]) {
  const int lane = threadIdx.x & 31;
  const int r    = lane & 15;
  const int half = lane >> 4;
#pragma unroll 2
  for (int k0 = 0; k0 < K; k0 += 32) {
    v16bf a[MT], b[NT];
#pragma unroll
    for (int i = 0; i < MT; ++i) a[i] = load_frag(A, m0 + i * 16 + r, k0, ldA, half);
#pragma unroll
    for (int j = 0; j < NT; ++j) b[j] = load_frag(W, n0 + j * 16 + r, k0, ldW, half);
#pragma unroll
    for (int i = 0; i < MT; ++i)
#pragma unroll
      for (int j = 0; j < NT; ++j)
        acc[i][j] = __builtin_amdgcn_wmma_f32_16x16x32_bf16(
            false, a[i], false, b[j], (short)0, acc[i][j], false, false);
  }
}

__device__ __forceinline__ void zero_acc(v8f acc[2][4]) {
  const v8f z = {0.f, 0.f, 0.f, 0.f, 0.f, 0.f, 0.f, 0.f};
#pragma unroll
  for (int i = 0; i < 2; ++i)
#pragma unroll
    for (int j = 0; j < 4; ++j) acc[i][j] = z;
}

// ---------------- helpers ----------------
__global__ void cvt_f32_bf16(const float* __restrict__ in, bf16_t* __restrict__ out, int n) {
  int i = blockIdx.x * blockDim.x + threadIdx.x;
  if (i < n) out[i] = (bf16_t)in[i];
}

__global__ void zero_bf16(bf16_t* __restrict__ out, int n) {
  int i = blockIdx.x * blockDim.x + threadIdx.x;
  if (i < n) out[i] = (bf16_t)0.0f;
}

__global__ void copy_f32(const float* __restrict__ src, float* __restrict__ dst, int n) {
  int i = blockIdx.x * blockDim.x + threadIdx.x;
  if (i < n) dst[i] = src[i];
}

// ---------------- proj = embed @ Wxh^T + Wxh_b + Whh_b  [V, H] ----------------
__global__ __launch_bounds__(256) void proj_kernel(const bf16_t* __restrict__ embed_bf,
                                                   const bf16_t* __restrict__ Wxh_bf,
                                                   const float* __restrict__ Wxh_b,
                                                   const float* __restrict__ Whh_b,
                                                   float* __restrict__ proj) {
  const int wave = threadIdx.x >> 5, lane = threadIdx.x & 31;
  const int r = lane & 15, half = lane >> 4;
  const int m0 = blockIdx.y * 64 + (wave & 1) * 32;   // vocab rows
  const int n0 = blockIdx.x * 256 + (wave >> 1) * 64; // hidden cols
  v8f acc[2][4];
  zero_acc(acc);
  gemm_bf16<2, 4>(embed_bf, H_, Wxh_bf, H_, H_, m0, n0, acc);
#pragma unroll
  for (int i = 0; i < 2; ++i)
#pragma unroll
    for (int j = 0; j < 4; ++j)
#pragma unroll
      for (int e = 0; e < 8; ++e) {
        int m = m0 + i * 16 + e + half * 8;
        int n = n0 + j * 16 + r;
        proj[m * H_ + n] = acc[i][j][e] + Wxh_b[n] + Whh_b[n];
      }
}

// ------------- h_t = tanh(proj[x[:,t]] + h_{t-1} @ Whh^T)  [B, H] -------------
__global__ __launch_bounds__(256) void step_kernel(const bf16_t* __restrict__ hprev,
                                                   const bf16_t* __restrict__ Whh_bf,
                                                   const float* __restrict__ proj,
                                                   const int* __restrict__ x, int t,
                                                   bf16_t* __restrict__ hnew_bf,
                                                   float* __restrict__ h32) {
  const int wave = threadIdx.x >> 5, lane = threadIdx.x & 31;
  const int r = lane & 15, half = lane >> 4;
  const int m0 = blockIdx.y * 64 + (wave & 1) * 32;   // batch rows
  const int n0 = blockIdx.x * 256 + (wave >> 1) * 64; // hidden cols
  v8f acc[2][4];
  zero_acc(acc);
  gemm_bf16<2, 4>(hprev, H_, Whh_bf, H_, H_, m0, n0, acc);
#pragma unroll
  for (int i = 0; i < 2; ++i) {
    // one x[] lookup per output row, reused across j/e
    int bi[8];
#pragma unroll
    for (int e = 0; e < 8; ++e) {
      int b = m0 + i * 16 + e + half * 8;
      bi[e] = x[b * S_ + t];
    }
#pragma unroll
    for (int j = 0; j < 4; ++j)
#pragma unroll
      for (int e = 0; e < 8; ++e) {
        int b = m0 + i * 16 + e + half * 8;
        int n = n0 + j * 16 + r;
        float val = tanhf(acc[i][j][e] + proj[bi[e] * H_ + n]);
        hnew_bf[b * H_ + n] = (bf16_t)val;
        h32[b * H_ + n] = val;
      }
  }
}

// ----- out[b, t0+tl, :] = hchunk[tl*B+b, :] @ fc_w^T + fc_b  (chunk of 32 t) -----
__global__ __launch_bounds__(256) void fc_kernel(const bf16_t* __restrict__ hchunk, // row = tl*B + b
                                                 const bf16_t* __restrict__ fcw_bf, // [V, H]
                                                 const float* __restrict__ fc_b,
                                                 float* __restrict__ out, int t0) {
  const int wave = threadIdx.x >> 5, lane = threadIdx.x & 31;
  const int r = lane & 15, half = lane >> 4;
  const int m0 = blockIdx.y * 256 + wave * 32; // (t,b) rows
  const int n0 = blockIdx.x * 64;              // vocab cols
  v8f acc[2][4];
  zero_acc(acc);
  gemm_bf16<2, 4>(hchunk, H_, fcw_bf, H_, H_, m0, n0, acc);
#pragma unroll
  for (int i = 0; i < 2; ++i)
#pragma unroll
    for (int j = 0; j < 4; ++j)
#pragma unroll
      for (int e = 0; e < 8; ++e) {
        int m = m0 + i * 16 + e + half * 8;
        int n = n0 + j * 16 + r;
        int tl = m >> 8;        // m / B_
        int b  = m & (B_ - 1);  // m % B_
        out[((size_t)b * S_ + (t0 + tl)) * V_ + n] = acc[i][j][e] + fc_b[n];
      }
}

extern "C" void kernel_launch(void* const* d_in, const int* in_sizes, int n_in,
                              void* d_out, int out_size, void* d_ws, size_t ws_size,
                              hipStream_t stream) {
  (void)in_sizes; (void)n_in; (void)out_size; (void)ws_size;
  const int*   x     = (const int*)d_in[0];
  const float* embed = (const float*)d_in[1];
  const float* Wxh_w = (const float*)d_in[2];
  const float* Wxh_b = (const float*)d_in[3];
  const float* Whh_w = (const float*)d_in[4];
  const float* Whh_b = (const float*)d_in[5];
  const float* fc_w  = (const float*)d_in[6];
  const float* fc_b  = (const float*)d_in[7];
  float* out = (float*)d_out;

  // workspace layout (~23.6 MB total)
  char* ws = (char*)d_ws;
  float*  proj     = (float*)(ws + 0);         // 128*1024*4   = 512 KB
  bf16_t* embed_bf = (bf16_t*)(ws + 524288);   // 256 KB
  bf16_t* Wxh_bf   = (bf16_t*)(ws + 786432);   // 2 MB
  bf16_t* Whh_bf   = (bf16_t*)(ws + 2883584);  // 2 MB
  bf16_t* fcw_bf   = (bf16_t*)(ws + 4980736);  // 256 KB
  bf16_t* hzero    = (bf16_t*)(ws + 5242880);  // 512 KB
  bf16_t* hchunk   = (bf16_t*)(ws + 5767168);  // 32*256*1024*2 = 16 MB ring
  float*  h32      = (float*)(ws + 22544384);  // 1 MB (last hidden, f32)

  cvt_f32_bf16<<<(V_ * H_ + 255) / 256, 256, 0, stream>>>(embed, embed_bf, V_ * H_);
  cvt_f32_bf16<<<(H_ * H_ + 255) / 256, 256, 0, stream>>>(Wxh_w, Wxh_bf, H_ * H_);
  cvt_f32_bf16<<<(H_ * H_ + 255) / 256, 256, 0, stream>>>(Whh_w, Whh_bf, H_ * H_);
  cvt_f32_bf16<<<(V_ * H_ + 255) / 256, 256, 0, stream>>>(fc_w, fcw_bf, V_ * H_);
  zero_bf16<<<(B_ * H_ + 255) / 256, 256, 0, stream>>>(hzero, B_ * H_);

  // proj[v,k] = sum_h embed[v,h]*Wxh[k,h] + Wxh_b[k] + Whh_b[k]
  proj_kernel<<<dim3(H_ / 256, V_ / 64), 256, 0, stream>>>(embed_bf, Wxh_bf, Wxh_b, Whh_b, proj);

  const int TC = 32; // chunk of timesteps buffered for the fc GEMM
  for (int t = 0; t < S_; ++t) {
    const bf16_t* hp = (t == 0) ? hzero
                                : (hchunk + (size_t)((t - 1) % TC) * B_ * H_);
    bf16_t* hn = hchunk + (size_t)(t % TC) * B_ * H_;
    step_kernel<<<dim3(H_ / 256, B_ / 64), 256, 0, stream>>>(hp, Whh_bf, proj, x, t, hn, h32);
    if ((t % TC) == TC - 1) {
      fc_kernel<<<dim3(V_ / 64, (TC * B_) / 256), 256, 0, stream>>>(hchunk, fcw_bf, fc_b, out,
                                                                    t - (TC - 1));
    }
  }
  // final hidden state -> second output
  copy_f32<<<(B_ * H_ + 255) / 256, 256, 0, stream>>>(h32, out + (size_t)B_ * S_ * V_, B_ * H_);
}
